// BondInfluenceSelfAttention_28587302322701
// MI455X (gfx1250) — matrix operand order repaired and checked by reading, not compiled, not run on hardware
//
#include <hip/hip_runtime.h>
#include <hip/hip_bf16.h>

#define DM    1024
#define HEADS 16
#define DKH   64
#define LSEQ  2048
#define NBATCH 2

typedef __attribute__((ext_vector_type(16))) __bf16 bf16x16;
typedef __attribute__((ext_vector_type(8)))  float  f32x8;

union Frag16 {           // one 16x32 A-frag or 32x16 B-frag of bf16 (8 VGPRs)
    bf16x16 v;
    unsigned short u[16];
    uint4 q[2];
};

__device__ __forceinline__ unsigned short f2bf(float f) {
    unsigned int x = __float_as_uint(f);
    x += 0x7FFFu + ((x >> 16) & 1u);     // round-to-nearest-even
    return (unsigned short)(x >> 16);
}

__device__ __forceinline__ f32x8 zero8() {
    f32x8 z = {0.f, 0.f, 0.f, 0.f, 0.f, 0.f, 0.f, 0.f};
    return z;
}

// ---------------------------------------------------------------------------
// Tiled GEMM with bias:  C[M,N] = A[M,K] @ W[K,N] + bias[N]
// Block tile 128x128, 8 waves (4x2), wave tile 32x64 (2x4 WMMA frags).
// A is fp32 or bf16(u16) source; W is fp32.
// OUT_MODE: 0 = fp32 [M,N], 1 = bf16 [M,N],
//           2 = bf16 transposed-per-head: Vt[((b*H+h)*64+dk)*LSEQ + l]
// ---------------------------------------------------------------------------
template <bool A_BF16, int OUT_MODE>
__global__ __launch_bounds__(256)
void gemm_bias_wmma(const void* __restrict__ Aptr,
                    const float* __restrict__ W,
                    const float* __restrict__ bias,
                    void* __restrict__ Cptr,
                    int M, int N, int K) {
    __shared__ unsigned short As[128][40];   // [m][k] bf16, 80B row stride
    __shared__ unsigned short Bt[128][40];   // [n][k] bf16 (W transposed)

    const int tid  = threadIdx.x;
    const int wid  = tid >> 5;
    const int lane = tid & 31;
    const int l15  = lane & 15;
    const int half = lane >> 4;
    const int wm   = wid >> 1;               // 0..3
    const int wn   = wid & 1;                // 0..1
    const int mblk = blockIdx.y * 128;
    const int nblk = blockIdx.x * 128;

    f32x8 acc[2][4];
#pragma unroll
    for (int i = 0; i < 2; ++i)
#pragma unroll
        for (int j = 0; j < 4; ++j) acc[i][j] = zero8();

    for (int kb = 0; kb < K; kb += 32) {
        // ---- stage A tile (128x32) and W tile transposed (32x128 -> [n][k])
#pragma unroll
        for (int it = 0; it < 4; ++it) {
            int lin = it * 1024 + tid * 4;
            int ar = lin >> 5, ac = lin & 31;            // A: row 0..127, col 0..31
            if constexpr (A_BF16) {
                const unsigned short* A16 = (const unsigned short*)Aptr;
#pragma unroll
                for (int j = 0; j < 4; ++j)
                    As[ar][ac + j] = A16[(size_t)(mblk + ar) * K + kb + ac + j];
            } else {
                const float* A32 = (const float*)Aptr;
#pragma unroll
                for (int j = 0; j < 4; ++j)
                    As[ar][ac + j] = f2bf(A32[(size_t)(mblk + ar) * K + kb + ac + j]);
            }
            int rk = lin >> 7, cn = lin & 127;           // W: k 0..31, n 0..127
#pragma unroll
            for (int j = 0; j < 4; ++j)
                Bt[cn + j][rk] = f2bf(W[(size_t)(kb + rk) * N + nblk + cn + j]);
        }
        __syncthreads();

        // ---- fragments per ISA layout
        Frag16 a[2], b[4];
#pragma unroll
        for (int mt = 0; mt < 2; ++mt) {
            int row = wm * 32 + mt * 16 + l15;
            a[mt].q[0] = *(const uint4*)&As[row][half * 8];
            a[mt].q[1] = *(const uint4*)&As[row][16 + half * 8];
        }
#pragma unroll
        for (int nt = 0; nt < 4; ++nt) {
            int col = wn * 64 + nt * 16 + l15;
            b[nt].q[0] = *(const uint4*)&Bt[col][half * 8];
            b[nt].q[1] = *(const uint4*)&Bt[col][16 + half * 8];
        }
#pragma unroll
        for (int mt = 0; mt < 2; ++mt)
#pragma unroll
            for (int nt = 0; nt < 4; ++nt)
                acc[mt][nt] = __builtin_amdgcn_wmma_f32_16x16x32_bf16(
                    false, a[mt].v, false, b[nt].v, (short)0, acc[mt][nt], false, false);
        __syncthreads();
    }

    // ---- epilogue: bias + store (C/D layout: lane=col, row = r + 8*half)
#pragma unroll
    for (int mt = 0; mt < 2; ++mt)
#pragma unroll
        for (int nt = 0; nt < 4; ++nt) {
            int n = nblk + wn * 64 + nt * 16 + l15;
            float bv = bias[n];
#pragma unroll
            for (int r = 0; r < 8; ++r) {
                int m = mblk + wm * 32 + mt * 16 + r + half * 8;
                float v = acc[mt][nt][r] + bv;
                if constexpr (OUT_MODE == 0) {
                    ((float*)Cptr)[(size_t)m * N + n] = v;
                } else if constexpr (OUT_MODE == 1) {
                    ((unsigned short*)Cptr)[(size_t)m * N + n] = f2bf(v);
                } else {
                    // transposed-per-head V layout: [b][h][dk][l]
                    int bb = m >> 11, l = m & (LSEQ - 1);
                    int hh = n >> 6,  dk = n & (DKH - 1);
                    size_t idx = (((size_t)(bb * HEADS + hh) * DKH + dk) * LSEQ) + l;
                    ((unsigned short*)Cptr)[idx] = f2bf(v);
                }
            }
        }
}

// ---------------------------------------------------------------------------
// Flash attention with bond modulation. One wave handles one (b,h,16-query)
// tile; loops over keys in blocks of 32. Q/K are bf16 in [B*L, DM] layout
// (== [B,L,H,dk] flat); V is bf16 transposed [B,H,dk,L]. O written bf16
// in [B*L, DM] layout. Softmax runs in the exp2 domain (v_exp_f32 == 2^x);
// row-sums are computed with a ones-matrix WMMA instead of lane shuffles.
// ---------------------------------------------------------------------------
__global__ __launch_bounds__(256)
void attn_bond_wmma(const unsigned short* __restrict__ Q,
                    const unsigned short* __restrict__ Kw,
                    const unsigned short* __restrict__ Vt,
                    const float* __restrict__ bond,
                    unsigned short* __restrict__ O) {
    __shared__ unsigned short Pl[8][16][40];  // per-wave 16x32 P tile, padded

    const int tid   = threadIdx.x;
    const int wslot = tid >> 5;
    const int lane  = tid & 31;
    const int l15   = lane & 15;
    const int half  = lane >> 4;

    const int wgid  = blockIdx.x * 8 + wslot;   // 0..4095
    const int qt    = wgid & 127;
    const int bh    = wgid >> 7;
    const int h     = bh & (HEADS - 1);
    const int b     = bh >> 4;
    const int qbase = qt * 16;

    const size_t batoff  = (size_t)b * LSEQ * DM;
    const size_t headoff = (size_t)h * DKH;
    const float* bondb   = bond + (size_t)b * LSEQ * LSEQ;
    // 1/sqrt(64) * log2(e): scores live in the exp2 domain
    const float  scale   = 0.125f * 1.44269504088896f;

    // Q A-fragments: 2 chunks over dk (K-dim 32 each), resident for the loop
    Frag16 qf[2];
#pragma unroll
    for (int c = 0; c < 2; ++c) {
        size_t base = batoff + (size_t)(qbase + l15) * DM + headoff + c * 32 + half * 8;
        qf[c].q[0] = *(const uint4*)&Q[base];
        qf[c].q[1] = *(const uint4*)&Q[base + 16];
    }

    // all-ones bf16 B-fragment for row-sum WMMA
    Frag16 ones;
#pragma unroll
    for (int e = 0; e < 16; ++e) ones.u[e] = 0x3F80;   // bf16 1.0

    f32x8 o[4];
#pragma unroll
    for (int i = 0; i < 4; ++i) o[i] = zero8();
    float mrow[8], lrow[8];
#pragma unroll
    for (int r = 0; r < 8; ++r) { mrow[r] = -1e30f; lrow[r] = 0.f; }

    for (int kb = 0; kb < LSEQ; kb += 32) {
        // prefetch next block's bond rows into cache (global_prefetch_b8)
        if (kb + 32 < LSEQ) {
#pragma unroll
            for (int r = 0; r < 8; ++r) {
                int q = qbase + r + half * 8;
                __builtin_prefetch(&bondb[(size_t)q * LSEQ + kb + 32 + l15], 0, 3);
            }
        }

        // ---- S = Q @ K^T for two 16-key subtiles
        f32x8 s[2];
#pragma unroll
        for (int t = 0; t < 2; ++t) {
            f32x8 cz = zero8();
#pragma unroll
            for (int c = 0; c < 2; ++c) {
                Frag16 kf;                       // B-frag: lane=key col, k=dk
                int key = kb + t * 16 + l15;
                size_t base = batoff + (size_t)key * DM + headoff + c * 32 + half * 8;
                kf.q[0] = *(const uint4*)&Kw[base];
                kf.q[1] = *(const uint4*)&Kw[base + 16];
                cz = __builtin_amdgcn_wmma_f32_16x16x32_bf16(
                    false, qf[c].v, false, kf.v, (short)0, cz, false, false);
            }
            s[t] = cz;
        }

        // ---- bond modulation (exp2 domain) + block row-max
        float bmax[8];
#pragma unroll
        for (int r = 0; r < 8; ++r) bmax[r] = -1e30f;
#pragma unroll
        for (int t = 0; t < 2; ++t) {
            int key = kb + t * 16 + l15;
#pragma unroll
            for (int r = 0; r < 8; ++r) {
                int q = qbase + r + half * 8;
                float v = s[t][r] * (scale * bondb[(size_t)q * LSEQ + key]);
                s[t][r] = v;
                bmax[r] = fmaxf(bmax[r], v);
            }
        }
#pragma unroll
        for (int r = 0; r < 8; ++r) {            // reduce over 16 lanes in half
            float v = bmax[r];
#pragma unroll
            for (int m = 1; m < 16; m <<= 1) v = fmaxf(v, __shfl_xor(v, m, 32));
            bmax[r] = v;
        }

        // ---- online softmax update (2^x throughout)
        float alpha[8];
#pragma unroll
        for (int r = 0; r < 8; ++r) {
            float mnew = fmaxf(mrow[r], bmax[r]);
            alpha[r] = __builtin_amdgcn_exp2f(mrow[r] - mnew);
            mrow[r] = mnew;
        }
#pragma unroll
        for (int t = 0; t < 2; ++t)
#pragma unroll
            for (int r = 0; r < 8; ++r)
                s[t][r] = __builtin_amdgcn_exp2f(s[t][r] - mrow[r]);
#pragma unroll
        for (int i = 0; i < 4; ++i)
#pragma unroll
            for (int r = 0; r < 8; ++r) o[i][r] = o[i][r] * alpha[r];

        // ---- transpose P through LDS: C-layout -> A-frag layout
#pragma unroll
        for (int t = 0; t < 2; ++t)
#pragma unroll
            for (int r = 0; r < 8; ++r)
                Pl[wslot][r + half * 8][t * 16 + l15] = f2bf(s[t][r]);
        __builtin_amdgcn_wave_barrier();
        asm volatile("s_wait_dscnt 0" ::: "memory");
        __builtin_amdgcn_wave_barrier();

        Frag16 pf;                               // A-frag: lane=query row, k=key
        pf.q[0] = *(const uint4*)&Pl[wslot][l15][half * 8];
        pf.q[1] = *(const uint4*)&Pl[wslot][l15][16 + half * 8];

        // ---- row-sum of P via ones-WMMA: every lane gets its 8 rows' sums
        f32x8 rs = __builtin_amdgcn_wmma_f32_16x16x32_bf16(
            false, pf.v, false, ones.v, (short)0, zero8(), false, false);
#pragma unroll
        for (int r = 0; r < 8; ++r) lrow[r] = lrow[r] * alpha[r] + rs[r];

        // ---- O += P @ V (4 dk subtiles); V is [b][h][dk][key] so the B-frag
        //      (lane = dk column, k = key) reads 2 contiguous b128 chunks
#pragma unroll
        for (int nt = 0; nt < 4; ++nt) {
            Frag16 vf;
            int dk = nt * 16 + l15;
            size_t vbase = (((size_t)(b * HEADS + h) * DKH + dk) * LSEQ) + kb + half * 8;
            vf.q[0] = *(const uint4*)&Vt[vbase];
            vf.q[1] = *(const uint4*)&Vt[vbase + 16];
            o[nt] = __builtin_amdgcn_wmma_f32_16x16x32_bf16(
                false, pf.v, false, vf.v, (short)0, o[nt], false, false);
        }
    }

    // ---- finalize: divide by l, store bf16
#pragma unroll
    for (int nt = 0; nt < 4; ++nt) {
        int dk = nt * 16 + l15;
#pragma unroll
        for (int r = 0; r < 8; ++r) {
            int q = qbase + r + half * 8;
            float v = o[nt][r] / lrow[r];
            O[batoff + (size_t)q * DM + headoff + dk] = f2bf(v);
        }
    }
}

// ---------------------------------------------------------------------------
extern "C" void kernel_launch(void* const* d_in, const int* in_sizes, int n_in,
                              void* d_out, int out_size, void* d_ws, size_t ws_size,
                              hipStream_t stream) {
    const float* x    = (const float*)d_in[0];
    const float* bond = (const float*)d_in[1];
    const float* Wq   = (const float*)d_in[2];
    const float* bq   = (const float*)d_in[3];
    const float* Wk   = (const float*)d_in[4];
    const float* bk   = (const float*)d_in[5];
    const float* Wv   = (const float*)d_in[6];
    const float* bv   = (const float*)d_in[7];
    const float* Wo   = (const float*)d_in[8];
    const float* bo   = (const float*)d_in[9];
    float* out = (float*)d_out;

    const int M = NBATCH * LSEQ;   // 4096
    const int N = DM;              // 1024
    const int K = DM;              // 1024

    const size_t elems = (size_t)M * DM;            // 4M bf16 each
    unsigned short* Qw = (unsigned short*)d_ws;     // 8 MB
    unsigned short* Kw = Qw + elems;                // 8 MB
    unsigned short* Vt = Kw + elems;                // 8 MB (transposed per head)
    unsigned short* Ow = Vt + elems;                // 8 MB

    dim3 gg(N / 128, M / 128);
    dim3 blk(256);

    gemm_bias_wmma<false, 1><<<gg, blk, 0, stream>>>(x, Wq, bq, Qw, M, N, K);
    gemm_bias_wmma<false, 1><<<gg, blk, 0, stream>>>(x, Wk, bk, Kw, M, N, K);
    gemm_bias_wmma<false, 2><<<gg, blk, 0, stream>>>(x, Wv, bv, Vt, M, N, K);

    // 4096 wave-tiles (= B*H*L/16), 8 waves per block
    attn_bond_wmma<<<512, blk, 0, stream>>>(Qw, Kw, Vt, bond, Ow);

    gemm_bias_wmma<true, 0><<<gg, blk, 0, stream>>>(Ow, Wo, bo, out, M, N, K);
}
